// VQCodebook_15917148799897
// MI455X (gfx1250) — matrix operand order repaired and checked by reading, not compiled, not run on hardware
//
#include <hip/hip_runtime.h>
#include <hip/hip_bf16.h>

typedef __bf16 bf16_t;
typedef __attribute__((ext_vector_type(16))) __bf16 v16bf;
typedef __attribute__((ext_vector_type(8)))  float  v8f;
typedef __attribute__((ext_vector_type(4)))  int    v4i;

#define BN_TOTAL 32768   // B*N
#define D_DIM    256
#define K_DIM    1024
#define TILE_M   128
#define TILE_N   128
#define KC       32      // D-chunk per LDS stage
#define LSTR     40      // LDS row stride in bf16 (32 + 8 pad = 80B: 16B-aligned, bank-friendly)
#define TOPM     3
#define EPSV     1e-7f

#if __has_builtin(__builtin_amdgcn_global_load_async_to_lds_b128) && \
    __has_builtin(__builtin_amdgcn_s_wait_asynccnt)
#define USE_ASYNC 1
#else
#define USE_ASYNC 0
#endif

#if USE_ASYNC
typedef __attribute__((address_space(1))) v4i gv4i;   // global int4
typedef __attribute__((address_space(3))) v4i lv4i;   // LDS int4
#endif

// ---------------------------------------------------------------------------
// Prep: split f32 q/codes into bf16 hi/lo arrays in workspace (one pass).
// hi = RNE(x); lo = RNE(x - float(hi))  ->  ~16 mantissa bits recovered.
// ---------------------------------------------------------------------------
#define Q_ELEMS  ((size_t)BN_TOTAL * D_DIM)   // 8388608
#define C_ELEMS  ((size_t)K_DIM * D_DIM)      // 262144

__global__ __launch_bounds__(256) void vq_split_kernel(const float* __restrict__ q,
                                                       const float* __restrict__ codes,
                                                       bf16_t* __restrict__ qh,
                                                       bf16_t* __restrict__ ql,
                                                       bf16_t* __restrict__ ch,
                                                       bf16_t* __restrict__ cl) {
  const size_t tid = (size_t)blockIdx.x * blockDim.x + threadIdx.x;  // 8 floats each
  const float* src;
  bf16_t *dh, *dl;
  size_t off;
  if (tid < Q_ELEMS / 8) {
    off = tid * 8;
    src = q + off; dh = qh + off; dl = ql + off;
  } else {
    off = (tid - Q_ELEMS / 8) * 8;
    if (off >= C_ELEMS) return;
    src = codes + off; dh = ch + off; dl = cl + off;
  }
  float xs[8];
  ((float4*)xs)[0] = ((const float4*)src)[0];
  ((float4*)xs)[1] = ((const float4*)src)[1];
  bf16_t hi[8], lo[8];
#pragma unroll
  for (int j = 0; j < 8; ++j) {
    bf16_t hb = (bf16_t)xs[j];
    hi[j] = hb;
    lo[j] = (bf16_t)(xs[j] - (float)hb);
  }
  *(uint4*)dh = *(uint4*)hi;
  *(uint4*)dl = *(uint4*)lo;
}

// ---------------------------------------------------------------------------
// GEMM: sim = q @ codes^T via 3-product bf16 WMMA (hi*hi + hi*lo + lo*hi),
// double-buffered LDS fed by async global->LDS copies (no VGPR round-trip).
// ---------------------------------------------------------------------------
__global__ __launch_bounds__(256) void vq_gemm_kernel(const bf16_t* __restrict__ qh,
                                                      const bf16_t* __restrict__ ql,
                                                      const bf16_t* __restrict__ ch,
                                                      const bf16_t* __restrict__ cl,
                                                      float* __restrict__ sim) {
  __shared__ bf16_t sQh[2][TILE_M * LSTR];
  __shared__ bf16_t sQl[2][TILE_M * LSTR];
  __shared__ bf16_t sCh[2][TILE_N * LSTR];
  __shared__ bf16_t sCl[2][TILE_N * LSTR];

  const int t       = threadIdx.x;
  const int rowBase = blockIdx.y * TILE_M;   // over B*N
  const int colBase = blockIdx.x * TILE_N;   // over K codes

  const int wave = t >> 5;
  const int lane = t & 31;
  const int mRow = wave * 16;
  const int lmod = lane & 15;
  const int lhi  = lane >> 4;

  // staging: thread t copies 32B (16 bf16) = half a KC-row of each array
  const int r = t >> 1;     // 0..127
  const int h = t & 1;      // half

  const size_t qOff = (size_t)(rowBase + r) * D_DIM + h * 16;
  const size_t cOff = (size_t)(colBase + r) * D_DIM + h * 16;
  const int    lOff = r * LSTR + h * 16;    // bf16 elems into LDS row

  v8f acc[8] = {};

  // ---- stage one KC chunk (32 bf16 cols) of all four arrays into buffer b ----
  auto stage = [&](int kc, int b) {
#if USE_ASYNC
    const bf16_t* gq0 = qh + qOff + kc;
    const bf16_t* gq1 = ql + qOff + kc;
    const bf16_t* gc0 = ch + cOff + kc;
    const bf16_t* gc1 = cl + cOff + kc;
    bf16_t* dq0 = &sQh[b][lOff];
    bf16_t* dq1 = &sQl[b][lOff];
    bf16_t* dc0 = &sCh[b][lOff];
    bf16_t* dc1 = &sCl[b][lOff];
    __builtin_amdgcn_global_load_async_to_lds_b128((gv4i*)gq0, (lv4i*)dq0, 0, 0);
    __builtin_amdgcn_global_load_async_to_lds_b128((gv4i*)gq0, (lv4i*)dq0, 16, 0);
    __builtin_amdgcn_global_load_async_to_lds_b128((gv4i*)gq1, (lv4i*)dq1, 0, 0);
    __builtin_amdgcn_global_load_async_to_lds_b128((gv4i*)gq1, (lv4i*)dq1, 16, 0);
    __builtin_amdgcn_global_load_async_to_lds_b128((gv4i*)gc0, (lv4i*)dc0, 0, 0);
    __builtin_amdgcn_global_load_async_to_lds_b128((gv4i*)gc0, (lv4i*)dc0, 16, 0);
    __builtin_amdgcn_global_load_async_to_lds_b128((gv4i*)gc1, (lv4i*)dc1, 0, 0);
    __builtin_amdgcn_global_load_async_to_lds_b128((gv4i*)gc1, (lv4i*)dc1, 16, 0);
#else
    uint4 a0 = ((const uint4*)(qh + qOff + kc))[0];
    uint4 a1 = ((const uint4*)(qh + qOff + kc))[1];
    uint4 b0 = ((const uint4*)(ql + qOff + kc))[0];
    uint4 b1 = ((const uint4*)(ql + qOff + kc))[1];
    uint4 c0 = ((const uint4*)(ch + cOff + kc))[0];
    uint4 c1 = ((const uint4*)(ch + cOff + kc))[1];
    uint4 d0 = ((const uint4*)(cl + cOff + kc))[0];
    uint4 d1 = ((const uint4*)(cl + cOff + kc))[1];
    ((uint4*)&sQh[b][lOff])[0] = a0;  ((uint4*)&sQh[b][lOff])[1] = a1;
    ((uint4*)&sQl[b][lOff])[0] = b0;  ((uint4*)&sQl[b][lOff])[1] = b1;
    ((uint4*)&sCh[b][lOff])[0] = c0;  ((uint4*)&sCh[b][lOff])[1] = c1;
    ((uint4*)&sCl[b][lOff])[0] = d0;  ((uint4*)&sCl[b][lOff])[1] = d1;
#endif
  };

  stage(0, 0);                       // prologue: chunk 0 -> buffer 0

  const int NCHUNK = D_DIM / KC;     // 8
  for (int i = 0; i < NCHUNK; ++i) {
    const int cur = i & 1;
    if (i + 1 < NCHUNK) {
      stage((i + 1) * KC, (i + 1) & 1);   // prefetch next chunk into other buffer
#if USE_ASYNC
      __builtin_amdgcn_s_wait_asynccnt(8);  // current chunk's 8 copies complete
#endif
    } else {
#if USE_ASYNC
      __builtin_amdgcn_s_wait_asynccnt(0);
#endif
    }
    __syncthreads();   // all waves' copies for `cur` are visible

    // ---- A fragments (16x32 bf16 per ISA layout) ----
    v16bf aHi, aLo;
    {
      const bf16_t* pa = &sQh[cur][(mRow + lmod) * LSTR + lhi * 8];
      ((uint4*)&aHi)[0] = *(const uint4*)pa;
      ((uint4*)&aHi)[1] = *(const uint4*)(pa + 16);
      const bf16_t* pl = &sQl[cur][(mRow + lmod) * LSTR + lhi * 8];
      ((uint4*)&aLo)[0] = *(const uint4*)pl;
      ((uint4*)&aLo)[1] = *(const uint4*)(pl + 16);
    }
#pragma unroll
    for (int n = 0; n < 8; ++n) {
      v16bf bHi, bLo;
      const bf16_t* pb = &sCh[cur][(n * 16 + lmod) * LSTR + lhi * 8];
      ((uint4*)&bHi)[0] = ((const uint4*)pb)[0];
      ((uint4*)&bHi)[1] = ((const uint4*)pb)[2];
      const bf16_t* pq = &sCl[cur][(n * 16 + lmod) * LSTR + lhi * 8];
      ((uint4*)&bLo)[0] = ((const uint4*)pq)[0];
      ((uint4*)&bLo)[1] = ((const uint4*)pq)[2];

      acc[n] = __builtin_amdgcn_wmma_f32_16x16x32_bf16(false, aHi, false, bHi,
                                                       (short)0, acc[n], false, false);
      acc[n] = __builtin_amdgcn_wmma_f32_16x16x32_bf16(false, aHi, false, bLo,
                                                       (short)0, acc[n], false, false);
      acc[n] = __builtin_amdgcn_wmma_f32_16x16x32_bf16(false, aLo, false, bHi,
                                                       (short)0, acc[n], false, false);
    }
    __syncthreads();   // done reading `cur`; its buffer may be overwritten next iter
  }

  // ---- store: C/D layout: lane L, VGPR r -> M = r + 8*(L/16), N = L%16 ----
  const int outRow0 = rowBase + mRow + lhi * 8;
#pragma unroll
  for (int n = 0; n < 8; ++n) {
    float* o = sim + (size_t)outRow0 * K_DIM + colBase + n * 16 + lmod;
#pragma unroll
    for (int rr = 0; rr < 8; ++rr) {
      o[(size_t)rr * K_DIM] = acc[n][rr];
    }
  }
}

// ---------------------------------------------------------------------------
// Top-3 + Gumbel softmax: one wave32 per (b,n) row.
// ---------------------------------------------------------------------------
__device__ __forceinline__ void top3_insert(float v, int i,
                                            float& v0, float& v1, float& v2,
                                            int& i0, int& i1, int& i2) {
  if (v > v2) {
    if (v > v0)      { v2 = v1; i2 = i1; v1 = v0; i1 = i0; v0 = v; i0 = i; }
    else if (v > v1) { v2 = v1; i2 = i1; v1 = v;  i1 = i; }
    else             { v2 = v;  i2 = i; }
  }
}

__global__ __launch_bounds__(256) void vq_topk_kernel(const float* __restrict__ sim,
                                                      const int* __restrict__ mask,
                                                      const float* __restrict__ u,
                                                      float* __restrict__ outIdx,
                                                      float* __restrict__ outW) {
  const int row  = blockIdx.x * 8 + threadIdx.y;
  const int lane = threadIdx.x;

  const int m = mask[row];
  if (m == 0) {
    if (lane < TOPM) {
      outIdx[row * TOPM + lane] = (float)lane;
      outW[row * TOPM + lane]   = 0.0f;
    }
    return;
  }

  float v0 = -3.0e38f, v1 = -3.0e38f, v2 = -3.0e38f;
  int   i0 = 0, i1 = 0, i2 = 0;
  const float* s = sim + (size_t)row * K_DIM;
#pragma unroll 4
  for (int j = 0; j < K_DIM / 32; ++j) {
    const int idx = lane + j * 32;
    top3_insert(s[idx], idx, v0, v1, v2, i0, i1, i2);
  }
#pragma unroll
  for (int off = 16; off > 0; off >>= 1) {
    float w0 = __shfl_xor(v0, off, 32);
    float w1 = __shfl_xor(v1, off, 32);
    float w2 = __shfl_xor(v2, off, 32);
    int   j0 = __shfl_xor(i0, off, 32);
    int   j1 = __shfl_xor(i1, off, 32);
    int   j2 = __shfl_xor(i2, off, 32);
    top3_insert(w0, j0, v0, v1, v2, i0, i1, i2);
    top3_insert(w1, j1, v0, v1, v2, i0, i1, i2);
    top3_insert(w2, j2, v0, v1, v2, i0, i1, i2);
  }

  if (lane == 0) {
    float vals[TOPM] = {v0, v1, v2};
    int   idxs[TOPM] = {i0, i1, i2};
    float l[TOPM];
    float mx = -3.0e38f;
#pragma unroll
    for (int j = 0; j < TOPM; ++j) {
      float uc = fminf(fmaxf(u[row * TOPM + j], EPSV), 1.0f - EPSV);
      float g  = -__logf(fmaxf(-__logf(uc), EPSV));
      l[j] = vals[j] + g;          // TAU = 1.0
      mx = fmaxf(mx, l[j]);
    }
    float e[TOPM], ssum = 0.0f;
#pragma unroll
    for (int j = 0; j < TOPM; ++j) { e[j] = __expf(l[j] - mx); ssum += e[j]; }
    const float inv = 1.0f / ssum;
#pragma unroll
    for (int j = 0; j < TOPM; ++j) {
      outIdx[row * TOPM + j] = (float)idxs[j];
      outW[row * TOPM + j]   = e[j] * inv;
    }
  }
}

// ---------------------------------------------------------------------------
extern "C" void kernel_launch(void* const* d_in, const int* in_sizes, int n_in,
                              void* d_out, int out_size, void* d_ws, size_t ws_size,
                              hipStream_t stream) {
  const float* q     = (const float*)d_in[0];   // [B,N,D] f32
  const float* codes = (const float*)d_in[1];   // [K,D]   f32
  const int*   mask  = (const int*)d_in[2];     // [B,N]   i32
  const float* u     = (const float*)d_in[3];   // [B,N,3] f32
  (void)in_sizes; (void)n_in; (void)out_size; (void)ws_size;

  float* outIdx = (float*)d_out;                       // [B,N,3]
  float* outW   = outIdx + (size_t)BN_TOTAL * TOPM;    // [B,N,3]
  float* sim    = outW + (size_t)BN_TOTAL * TOPM;      // [B,N,K]

  bf16_t* qh = (bf16_t*)d_ws;                          // [B*N, D] bf16 hi
  bf16_t* ql = qh + Q_ELEMS;                           // [B*N, D] bf16 lo
  bf16_t* ch = ql + Q_ELEMS;                           // [K, D]   bf16 hi
  bf16_t* cl = ch + C_ELEMS;                           // [K, D]   bf16 lo

  const int splitThreads = (int)((Q_ELEMS + C_ELEMS) / 8);
  vq_split_kernel<<<(splitThreads + 255) / 256, 256, 0, stream>>>(q, codes, qh, ql, ch, cl);

  dim3 gGemm(K_DIM / TILE_N, BN_TOTAL / TILE_M);       // (8, 256)
  vq_gemm_kernel<<<gGemm, 256, 0, stream>>>(qh, ql, ch, cl, sim);

  dim3 bTop(32, 8);
  vq_topk_kernel<<<BN_TOTAL / 8, bTop, 0, stream>>>(sim, mask, u, outIdx, outW);
}
